// MultiHeadAttention_8572754723468
// MI455X (gfx1250) — matrix operand order repaired and checked by reading, not compiled, not run on hardware
//
#include <hip/hip_runtime.h>
#include <hip/hip_bf16.h>

// ---------------- problem constants ----------------
#define H_  16
#define E_  1024
#define D_  64
#define M_  64
#define O_  1024
#define S_  2048
#define HM_ 1024   // H_*M_

typedef __bf16 bf16_t;
typedef __attribute__((ext_vector_type(16))) __bf16 v16bf;
typedef __attribute__((ext_vector_type(8)))  __bf16 v8bf;
typedef __attribute__((ext_vector_type(8)))  float  v8f;
typedef int v4i __attribute__((vector_size(16)));   // matches builtin's V4i param

#define AS1 __attribute__((address_space(1)))
#define AS3 __attribute__((address_space(3)))

#if __has_builtin(__builtin_amdgcn_global_load_async_to_lds_b128)
#define FLASH_ASYNC 1
#endif

// ---------------- fp32 -> bf16 convert ----------------
__global__ __launch_bounds__(256)
void cvt_bf16_kernel(const float* __restrict__ in, bf16_t* __restrict__ out, int n) {
  for (int i = blockIdx.x * blockDim.x + threadIdx.x; i < n; i += gridDim.x * blockDim.x)
    out[i] = (bf16_t)in[i];
}

// in: [B][R][C] fp32  ->  out: [B][C][R] bf16   (Bt layout so B-fragment loads are contiguous)
__global__ __launch_bounds__(256)
void transpose_cvt_kernel(const float* __restrict__ in, bf16_t* __restrict__ out,
                          int Bn, int R, int C) {
  size_t total = (size_t)Bn * R * C;
  for (size_t o = (size_t)blockIdx.x * blockDim.x + threadIdx.x; o < total;
       o += (size_t)gridDim.x * blockDim.x) {
    size_t rc  = (size_t)R * C;
    size_t b   = o / rc;
    size_t rem = o - b * rc;
    size_t c   = rem / R;
    size_t r   = rem - c * R;
    out[o] = (bf16_t)in[b * rc + r * C + c];
  }
}

// ---------------- fragment builders ----------------
// A fragment: 16 contiguous K values [p..p+15]
__device__ __forceinline__ v16bf fragA(const bf16_t* p) {
  v8bf lo = *(const v8bf*)p;
  v8bf hi = *(const v8bf*)(p + 8);
  v16bf r;
#pragma unroll
  for (int i = 0; i < 8; ++i) { r[i] = lo[i]; r[i + 8] = hi[i]; }
  return r;
}
// B fragment: 16 contiguous K values [p..p+15]
__device__ __forceinline__ v16bf fragB(const bf16_t* p) {
  v8bf lo = *(const v8bf*)p;
  v8bf hi = *(const v8bf*)(p + 8);
  v16bf r;
#pragma unroll
  for (int i = 0; i < 8; ++i) { r[i] = lo[i]; r[i + 8] = hi[i]; }
  return r;
}

// ---------------- generic WMMA GEMM (projections) ----------------
// MODE 0: bf16 row-major store (+bias)          (Q/K proj)
// MODE 2: bf16 TRANSPOSED store (+bias)         (V proj -> V^T)
// MODE 3: fp32 row-major store (+bias)          (final proj)
template <int MODE>
__global__ __launch_bounds__(128)
void wmma_gemm_kernel(const bf16_t* __restrict__ A, int lda, long strideA,
                      const bf16_t* __restrict__ Bt, int ldb, long strideBt,
                      int K,
                      const float* __restrict__ bias, long strideBias,
                      void* __restrict__ outv, int ldo, long strideOut, int colOff) {
  const int h = blockIdx.z;
  A  += (size_t)h * (size_t)strideA;
  Bt += (size_t)h * (size_t)strideBt;
  const float* bvec = bias ? bias + (size_t)h * (size_t)strideBias : nullptr;

  const int wave = threadIdx.x >> 5;
  const int lane = threadIdx.x & 31;
  const int l15  = lane & 15;
  const int laneHalf = lane >> 4;

  const int row0 = blockIdx.y * 64 + wave * 16;
  const int col0 = blockIdx.x * 64;

  v8f acc[4];
#pragma unroll
  for (int j = 0; j < 4; ++j)
#pragma unroll
    for (int i = 0; i < 8; ++i) acc[j][i] = 0.0f;

  const bf16_t* Arow = A + (size_t)(row0 + l15) * (size_t)lda;

  for (int kk = 0; kk < K; kk += 32) {
    v16bf a = fragA(Arow + kk + laneHalf * 16);
#pragma unroll
    for (int j = 0; j < 4; ++j) {
      const v16bf b =
          *(const v16bf*)(Bt + (size_t)(col0 + j * 16 + l15) * (size_t)ldb +
                          kk + laneHalf * 16);
      acc[j] = __builtin_amdgcn_wmma_f32_16x16x32_bf16(
          false, a, false, b, (short)0, acc[j], false, false);
    }
  }

  float*  outf = (float*)outv;
  bf16_t* outb = (bf16_t*)outv;
#pragma unroll
  for (int j = 0; j < 4; ++j) {
    const int c = col0 + j * 16 + l15;
#pragma unroll
    for (int jj = 0; jj < 8; ++jj) {
      const int r = row0 + jj + laneHalf * 8;
      float v = acc[j][jj];
      if (bvec) v += bvec[c];
      if (MODE == 0) {
        outb[(size_t)h * (size_t)strideOut + (size_t)r * ldo + colOff + c] = (bf16_t)v;
      } else if (MODE == 2) {
        outb[(size_t)h * (size_t)strideOut + (size_t)(colOff + c) * ldo + r] = (bf16_t)v;
      } else {
        outf[(size_t)r * ldo + c] = v;
      }
    }
  }
}

// ---------------- cross-lane reductions via ds_swizzle (xor within 16-lane group) ----------------
template <int IMM>
__device__ __forceinline__ float swz(float v) {
  return __int_as_float(__builtin_amdgcn_ds_swizzle(__float_as_int(v), IMM));
}
__device__ __forceinline__ float reduce16_max(float v) {
  v = fmaxf(v, swz<0x041f>(v));
  v = fmaxf(v, swz<0x081f>(v));
  v = fmaxf(v, swz<0x101f>(v));
  v = fmaxf(v, swz<0x201f>(v));
  return v;
}
__device__ __forceinline__ float reduce16_sum(float v) {
  v += swz<0x041f>(v);
  v += swz<0x081f>(v);
  v += swz<0x101f>(v);
  v += swz<0x201f>(v);
  return v;
}

// 16B global->LDS copy: async (GLOBAL_LOAD_ASYNC_TO_LDS_B128, ASYNCcnt) when available
__device__ __forceinline__ void copy16(const bf16_t* g, bf16_t* l) {
#ifdef FLASH_ASYNC
  __builtin_amdgcn_global_load_async_to_lds_b128(
      (AS1 v4i*)g, (AS3 v4i*)l, 0, 0);
#else
  *(v8bf*)l = *(const v8bf*)g;
#endif
}

#define LK 72  // padded LDS row stride (elements) for 64-wide tiles

// ---------------- fused flash attention ----------------
// grid = (S/64, H); block = 128 (4 waves). Wave w owns Q rows [blk*64+w*16, +16).
// K/V 64x64 tiles are double-buffered in LDS via async copies shared by all
// 4 waves; per tile: S = Q K^T (WMMA) -> online softmax (ds_swizzle row
// reductions) -> P via wave-private LDS transpose -> O += P V (WMMA).
__global__ __launch_bounds__(128)
void flash_attn_kernel(const bf16_t* __restrict__ Qb,  // [H][S][D]
                       const bf16_t* __restrict__ Kb,  // [H][T][D]
                       const bf16_t* __restrict__ Vt,  // [H][M][T]
                       const int*    __restrict__ mask,// [S][T]
                       bf16_t*       __restrict__ Oc)  // [S][HM]
{
  __shared__ bf16_t ldsK[2][64][LK];
  __shared__ bf16_t ldsV[2][64][LK];
  __shared__ bf16_t ldsP[4][16][LK];

  const int h    = blockIdx.y;
  const int tid  = threadIdx.x;
  const int wave = tid >> 5;
  const int lane = tid & 31;
  const int l15  = lane & 15;
  const int laneHalf = lane >> 4;
  const int row0 = blockIdx.x * 64 + wave * 16;

  const bf16_t* Qh = Qb + (size_t)h * S_ * D_;
  const bf16_t* Kh = Kb + (size_t)h * S_ * D_;
  const bf16_t* Vh = Vt + (size_t)h * M_ * S_;

  // stage one 64x64 K tile + 64x64 V tile into LDS buffer `buf`
  auto stageKV = [&](int t0, int buf) {
#pragma unroll
    for (int c = 0; c < 4; ++c) {
      const int chunk = tid + c * 128;       // 0..511
      const int row   = chunk >> 3;          // 0..63
      const int col8  = (chunk & 7) * 8;     // element column (x8)
      copy16(Kh + (size_t)(t0 + row) * D_ + col8, &ldsK[buf][row][col8]);
    }
#pragma unroll
    for (int c = 0; c < 4; ++c) {
      const int chunk = tid + c * 128;
      const int row   = chunk >> 3;
      const int col8  = (chunk & 7) * 8;
      copy16(Vh + (size_t)row * S_ + t0 + col8, &ldsV[buf][row][col8]);
    }
  };

  // Preload this wave's Q A-fragments (rows row0..row0+15, all 64 of D)
  v16bf aq[2];
#pragma unroll
  for (int ks = 0; ks < 2; ++ks)
    aq[ks] = fragA(Qh + (size_t)(row0 + l15) * D_ + ks * 32 + laneHalf * 16);

  v8f oacc[4];
#pragma unroll
  for (int j = 0; j < 4; ++j)
#pragma unroll
    for (int i = 0; i < 8; ++i) oacc[j][i] = 0.0f;

  float mrow[8], lrow[8];
#pragma unroll
  for (int jj = 0; jj < 8; ++jj) { mrow[jj] = -__builtin_inff(); lrow[jj] = 0.0f; }

  stageKV(0, 0);

  for (int tt = 0; tt < S_ / 64; ++tt) {
    const int t0  = tt * 64;
    const int cur = tt & 1;
    const bool haveNext = (t0 + 64 < S_);

    if (haveNext) stageKV(t0 + 64, cur ^ 1);   // double-buffer: issue next tile
#ifdef FLASH_ASYNC
    if (haveNext) asm volatile("s_wait_asynccnt 0x8" ::: "memory");
    else          asm volatile("s_wait_asynccnt 0x0" ::: "memory");
#endif
    __syncthreads();  // all waves' copies for `cur` visible

    // prefetch next mask tile rows (global_prefetch_b8)
    if (haveNext)
      __builtin_prefetch(mask + (size_t)(row0 + l15) * S_ + t0 + 64 + laneHalf * 32, 0, 1);

    // ---- S tile = Q K^T (K from LDS) ----
    v8f sacc[4];
#pragma unroll
    for (int j = 0; j < 4; ++j)
#pragma unroll
      for (int i = 0; i < 8; ++i) sacc[j][i] = 0.0f;

#pragma unroll
    for (int ks = 0; ks < 2; ++ks) {
#pragma unroll
      for (int j = 0; j < 4; ++j) {
        const v16bf b = fragB(&ldsK[cur][j * 16 + l15][ks * 32 + laneHalf * 16]);
        sacc[j] = __builtin_amdgcn_wmma_f32_16x16x32_bf16(
            false, aq[ks], false, b, (short)0, sacc[j], false, false);
      }
    }

    // ---- online softmax update (fp32), P kept in sacc ----
    float fac[8];
#pragma unroll
    for (int jj = 0; jj < 8; ++jj) {
      const int r = row0 + jj + laneHalf * 8;
      float chmax = -__builtin_inff();
#pragma unroll
      for (int j = 0; j < 4; ++j) {
        float v = sacc[j][jj] * 0.125f;  // 1/sqrt(D)
        const int c = t0 + j * 16 + l15;
        if (mask[(size_t)r * S_ + c] == 0) v = -__builtin_inff();
        sacc[j][jj] = v;
        chmax = fmaxf(chmax, v);
      }
      chmax = reduce16_max(chmax);
      const float mnew = fmaxf(mrow[jj], chmax);
      const bool dead = (mnew == -__builtin_inff());
      const float f = dead ? 1.0f : __expf(mrow[jj] - mnew);
      mrow[jj] = mnew;
      fac[jj] = f;
      float rsum = 0.0f;
#pragma unroll
      for (int j = 0; j < 4; ++j) {
        const float pv = dead ? 0.0f : __expf(sacc[j][jj] - mnew);
        sacc[j][jj] = pv;
        rsum += pv;
      }
      rsum = reduce16_sum(rsum);
      lrow[jj] = lrow[jj] * f + rsum;
    }
#pragma unroll
    for (int j = 0; j < 4; ++j)
#pragma unroll
      for (int jj = 0; jj < 8; ++jj) oacc[j][jj] *= fac[jj];

    // ---- stage P (bf16) to wave-private LDS (cross-lane transpose) ----
#pragma unroll
    for (int j = 0; j < 4; ++j)
#pragma unroll
      for (int jj = 0; jj < 8; ++jj)
        ldsP[wave][jj + laneHalf * 8][j * 16 + l15] = (bf16_t)sacc[j][jj];
    __syncthreads();

    // ---- O += P V  (A from ldsP, B from ldsV) ----
#pragma unroll
    for (int ks = 0; ks < 2; ++ks) {
      v16bf a = fragA(&ldsP[wave][l15][ks * 32 + laneHalf * 16]);
#pragma unroll
      for (int j = 0; j < 4; ++j) {
        const v16bf b = fragB(&ldsV[cur][j * 16 + l15][ks * 32 + laneHalf * 16]);
        oacc[j] = __builtin_amdgcn_wmma_f32_16x16x32_bf16(
            false, a, false, b, (short)0, oacc[j], false, false);
      }
    }
    __syncthreads();  // protect cur buffers before they are restaged
  }

  // ---- finalize: O / l -> concat buffer columns [h*64, h*64+64) ----
#pragma unroll
  for (int j = 0; j < 4; ++j) {
    const int c = h * M_ + j * 16 + l15;
#pragma unroll
    for (int jj = 0; jj < 8; ++jj) {
      const int r = row0 + jj + laneHalf * 8;
      Oc[(size_t)r * HM_ + c] = (bf16_t)(oacc[j][jj] / lrow[jj]);
    }
  }
}

// ---------------- host ----------------
extern "C" void kernel_launch(void* const* d_in, const int* in_sizes, int n_in,
                              void* d_out, int out_size, void* d_ws, size_t ws_size,
                              hipStream_t stream) {
  const float* x    = (const float*)d_in[0];
  const float* z    = (const float*)d_in[1];
  const int*   mask = (const int*)  d_in[2];
  const float* Wq   = (const float*)d_in[3];
  const float* bq   = (const float*)d_in[4];
  const float* Wk   = (const float*)d_in[5];
  const float* bk   = (const float*)d_in[6];
  const float* Wv   = (const float*)d_in[7];
  const float* bv   = (const float*)d_in[8];
  const float* W0   = (const float*)d_in[9];
  const float* b0   = (const float*)d_in[10];
  float* out = (float*)d_out;

  size_t off = 0;
  auto carve = [&](size_t bytes) { size_t o = off; off += (bytes + 255) & ~(size_t)255; return o; };
  char* ws = (char*)d_ws;
  bf16_t* xb  = (bf16_t*)(ws + carve((size_t)S_ * E_ * 2));
  bf16_t* zb  = (bf16_t*)(ws + carve((size_t)S_ * E_ * 2));
  bf16_t* WqT = (bf16_t*)(ws + carve((size_t)H_ * D_ * E_ * 2));  // [H][D][E]
  bf16_t* WkT = (bf16_t*)(ws + carve((size_t)H_ * D_ * E_ * 2));
  bf16_t* WvT = (bf16_t*)(ws + carve((size_t)H_ * M_ * E_ * 2));
  bf16_t* W0T = (bf16_t*)(ws + carve((size_t)O_ * HM_ * 2));      // [O][HM]
  bf16_t* Qb  = (bf16_t*)(ws + carve((size_t)H_ * S_ * D_ * 2));  // [H][S][D]
  bf16_t* Kb  = (bf16_t*)(ws + carve((size_t)H_ * S_ * D_ * 2));  // [H][T][D]
  bf16_t* Vt  = (bf16_t*)(ws + carve((size_t)H_ * M_ * S_ * 2));  // [H][M][T]
  bf16_t* Oc  = (bf16_t*)(ws + carve((size_t)S_ * HM_ * 2));      // concat [S][HM]

  cvt_bf16_kernel<<<dim3((S_ * E_ + 255) / 256), dim3(256), 0, stream>>>(x, xb, S_ * E_);
  cvt_bf16_kernel<<<dim3((S_ * E_ + 255) / 256), dim3(256), 0, stream>>>(z, zb, S_ * E_);

  {
    int nW = H_ * E_ * D_;
    transpose_cvt_kernel<<<dim3((nW + 255) / 256), dim3(256), 0, stream>>>(Wq, WqT, H_, E_, D_);
    transpose_cvt_kernel<<<dim3((nW + 255) / 256), dim3(256), 0, stream>>>(Wk, WkT, H_, E_, D_);
    transpose_cvt_kernel<<<dim3((nW + 255) / 256), dim3(256), 0, stream>>>(Wv, WvT, H_, E_, M_);
    int n0 = HM_ * O_;
    transpose_cvt_kernel<<<dim3((n0 + 255) / 256), dim3(256), 0, stream>>>(W0, W0T, 1, HM_, O_);
  }

  wmma_gemm_kernel<0><<<dim3(D_ / 64, S_ / 64, H_), dim3(128), 0, stream>>>(
      xb, E_, 0L, WqT, E_, (long)D_ * E_, E_, bq, (long)D_,
      Qb, D_, (long)S_ * D_, 0);
  wmma_gemm_kernel<0><<<dim3(D_ / 64, S_ / 64, H_), dim3(128), 0, stream>>>(
      zb, E_, 0L, WkT, E_, (long)D_ * E_, E_, bk, (long)D_,
      Kb, D_, (long)S_ * D_, 0);
  wmma_gemm_kernel<2><<<dim3(M_ / 64, S_ / 64, H_), dim3(128), 0, stream>>>(
      zb, E_, 0L, WvT, E_, (long)M_ * E_, E_, bv, (long)M_,
      Vt, S_, (long)M_ * S_, 0);

  flash_attn_kernel<<<dim3(S_ / 64, H_), dim3(128), 0, stream>>>(Qb, Kb, Vt, mask, Oc);

  wmma_gemm_kernel<3><<<dim3(O_ / 64, S_ / 64, 1), dim3(128), 0, stream>>>(
      Oc, HM_, 0L, W0T, HM_, 0L, HM_, b0, 0L,
      out, O_, 0L, 0);

  (void)in_sizes; (void)n_in; (void)out_size; (void)ws_size;
}
